// GMMSegHead_24696061952473
// MI455X (gfx1250) — matrix operand order repaired and checked by reading, not compiled, not run on hardware
//
#include <hip/hip_runtime.h>
#include <hip/hip_bf16.h>
#include <math.h>

typedef __attribute__((ext_vector_type(16))) _Float16 v16h;
typedef __attribute__((ext_vector_type(8)))  _Float16 v8h;
typedef __attribute__((ext_vector_type(8)))  float    v8f;

#define C_DIM 512
#define HWn   16384
#define KCLS  19
#define MCMP  5
#define PREAL 95
#define PPAD  96
#define TPX   32      // pixels per block
#define FPAD  520     // padded f16 row of f-tile (260 dwords -> stride 4 mod 64 banks)
#define SPAD  97      // padded f32 row of S-tile (odd stride -> conflict free)

// ---------------------------------------------------------------------------
// Kernel A: per-Gaussian weight prep.
//   mu = l2_normalize(means[p]); iv = 1/sigma^2
//   Wq2[p] = -2*mu*iv (f16), Wiv[p] = iv (f16)
//   bias[p] = -0.5*sum(mu^2*iv) - sum(log sigma) - 0.5*D*log(2pi)
// ---------------------------------------------------------------------------
__global__ void __launch_bounds__(128) gmm_prep(const float* __restrict__ means,
                                                const float* __restrict__ diag,
                                                _Float16* __restrict__ Wq2,
                                                _Float16* __restrict__ Wiv,
                                                float* __restrict__ bias) {
  const int p = blockIdx.x;
  const int tid = threadIdx.x;
  if (p >= PREAL) {  // pad row: zero so the GEMM column 95 is benign
    for (int c = tid; c < C_DIM; c += 128) {
      Wq2[(size_t)p * C_DIM + c] = (_Float16)0.0f;
      Wiv[(size_t)p * C_DIM + c] = (_Float16)0.0f;
    }
    if (tid == 0) bias[p] = 0.0f;
    return;
  }
  __shared__ float rbuf[128];
  __shared__ float sh_norm;
  float m[4], sg[4];
  float n2 = 0.0f;
  for (int i = 0; i < 4; ++i) {
    int c = tid + i * 128;
    m[i]  = means[(size_t)p * C_DIM + c];
    sg[i] = diag[(size_t)p * C_DIM + c];
    n2 += m[i] * m[i];
  }
  rbuf[tid] = n2;
  __syncthreads();
  if (tid == 0) {
    float s = 0.0f;
    for (int i = 0; i < 128; ++i) s += rbuf[i];
    sh_norm = sqrtf(s);
  }
  __syncthreads();
  const float dn = fmaxf(sh_norm, 1e-12f);
  float q3 = 0.0f, ld = 0.0f;
  for (int i = 0; i < 4; ++i) {
    int c = tid + i * 128;
    float mu = m[i] / dn;
    float iv = 1.0f / (sg[i] * sg[i]);
    Wq2[(size_t)p * C_DIM + c] = (_Float16)(-2.0f * mu * iv);
    Wiv[(size_t)p * C_DIM + c] = (_Float16)iv;
    q3 += mu * mu * iv;
    ld += logf(sg[i]);
  }
  __syncthreads();
  rbuf[tid] = q3;
  __syncthreads();
  float q3s = 0.0f;
  if (tid == 0) { for (int i = 0; i < 128; ++i) q3s += rbuf[i]; }
  __syncthreads();
  rbuf[tid] = ld;
  __syncthreads();
  if (tid == 0) {
    float lds = 0.0f;
    for (int i = 0; i < 128; ++i) lds += rbuf[i];
    const float HL2PI = 0.5f * (float)C_DIM * logf(6.28318530717958647692f);
    bias[p] = -0.5f * q3s - lds - HL2PI;
  }
}

// ---------------------------------------------------------------------------
// Kernel B: fused transpose + LayerNorm + L2 + WMMA Mahalanobis + max + LN.
// One block = 32 pixels, 128 threads (4 wave32).
// ---------------------------------------------------------------------------
__global__ void __launch_bounds__(128) gmm_main(
    const float* __restrict__ x,
    const float* __restrict__ fgamma, const float* __restrict__ fbeta,
    const float* __restrict__ mgamma, const float* __restrict__ mbeta,
    const _Float16* __restrict__ Wq2, const _Float16* __restrict__ Wiv,
    const float* __restrict__ bias,
    float* __restrict__ out) {
  __shared__ __align__(16) char smem_u[TPX * FPAD * 2];  // f-tile f16 / S-tile f32 overlay
  __shared__ float gam[C_DIM], bet[C_DIM];
  __shared__ float red[8][TPX];
  __shared__ float mu_s[TPX], is_s[TPX];

  _Float16* ftile = (_Float16*)smem_u;  // [32][FPAD] f16
  float*    Stile = (float*)smem_u;     // [32][SPAD] f32 (after GEMM)

  const int tid  = threadIdx.x;
  const int px   = tid & 31;
  const int part = tid >> 5;            // wave id = C-chunk id
  const int n0   = blockIdx.x * TPX;
  const int b    = n0 >> 14;            // pixel tile never crosses batch (16384 % 32 == 0)
  const int hw   = n0 & (HWn - 1);

  for (int i = tid; i < C_DIM; i += 128) { gam[i] = fgamma[i]; bet[i] = fbeta[i]; }

  // -- pass 1: global -> LDS (f16), accumulate per-pixel mean / var --------
  const float* xb = x + (size_t)b * C_DIM * HWn + hw + px;  // lane == px -> coalesced
  const int cb = part * 128;
  float s1 = 0.0f, s2 = 0.0f;
  for (int it = 0; it < 128; ++it) {
    int c = cb + it;
    float v = xb[(size_t)c * HWn];
    ftile[px * FPAD + c] = (_Float16)v;
    s1 += v;
    s2 += v * v;
  }
  red[part][px] = s1;
  red[4 + part][px] = s2;
  __syncthreads();
  if (tid < 32) {
    float su = red[0][tid] + red[1][tid] + red[2][tid] + red[3][tid];
    float sq = red[4][tid] + red[5][tid] + red[6][tid] + red[7][tid];
    float mean = su * (1.0f / C_DIM);
    float var  = sq * (1.0f / C_DIM) - mean * mean;
    mu_s[tid] = mean;
    is_s[tid] = rsqrtf(var + 1e-5f);
  }
  __syncthreads();

  // -- pass 2: LayerNorm in LDS, accumulate sum-of-squares for L2 ----------
  {
    const float mean = mu_s[px], is = is_s[px];
    float ss = 0.0f;
    for (int it = 0; it < 128; ++it) {
      int c = cb + it;
      float y = ((float)ftile[px * FPAD + c] - mean) * is * gam[c] + bet[c];
      ftile[px * FPAD + c] = (_Float16)y;
      ss += y * y;
    }
    red[part][px] = ss;
  }
  __syncthreads();
  if (tid < 32) {
    float ss = red[0][tid] + red[1][tid] + red[2][tid] + red[3][tid];
    mu_s[tid] = 1.0f / fmaxf(sqrtf(ss), 1e-12f);  // reuse mu_s as L2 scale
  }
  __syncthreads();

  // -- pass 3: scale rows to unit L2 norm ----------------------------------
  {
    const float r = mu_s[px];
    for (int it = 0; it < 128; ++it) {
      int c = cb + it;
      ftile[px * FPAD + c] = (_Float16)((float)ftile[px * FPAD + c] * r);
    }
  }
  __syncthreads();

  // -- WMMA GEMM: S[32x96] = sum_k f^2*iv - 2 f*mu*iv  (shared accumulator)
  const int lane  = tid & 31;
  const int mrow  = lane & 15;
  const int mhalf = lane >> 4;
  const int m0t   = part & 1;            // which 16-pixel M tile
  const int ng    = (part >> 1) * 3;     // first of 3 N tiles for this wave
  v8f acc0 = {}, acc1 = {}, acc2 = {};

  const int arow = (m0t * 16 + mrow) * FPAD;
  const size_t bstride = (size_t)16 * C_DIM;
  const _Float16* wqBase = Wq2 + (size_t)(ng * 16 + mrow) * C_DIM + mhalf * 16;
  const _Float16* wiBase = Wiv + (size_t)(ng * 16 + mrow) * C_DIM + mhalf * 16;

  for (int ks = 0; ks < 16; ++ks) {
    const int k0 = ks * 32;
    // A fragment (this code's contiguous K order, same as B): lane-half h holds 16 consecutive K {16h..16h+15}
    v8h alo = *(const v8h*)&ftile[arow + k0 + mhalf * 16];
    v8h ahi = *(const v8h*)&ftile[arow + k0 + mhalf * 16 + 8];
    v16h a, a2;
    for (int i = 0; i < 8; ++i) { a[i] = alo[i]; a[8 + i] = ahi[i]; }
    for (int i = 0; i < 16; ++i) a2[i] = a[i] * a[i];   // f^2 fragment, same layout
    // B fragments (32x16): lane n = lane&15 holds 16 consecutive K at (lane>>4)*16
    v16h b0q = *(const v16h*)(wqBase + k0);
    v16h b0i = *(const v16h*)(wiBase + k0);
    v16h b1q = *(const v16h*)(wqBase + bstride + k0);
    v16h b1i = *(const v16h*)(wiBase + bstride + k0);
    v16h b2q = *(const v16h*)(wqBase + 2 * bstride + k0);
    v16h b2i = *(const v16h*)(wiBase + 2 * bstride + k0);
    acc0 = __builtin_amdgcn_wmma_f32_16x16x32_f16(false, a,  false, b0q, (short)0, acc0, false, false);
    acc0 = __builtin_amdgcn_wmma_f32_16x16x32_f16(false, a2, false, b0i, (short)0, acc0, false, false);
    acc1 = __builtin_amdgcn_wmma_f32_16x16x32_f16(false, a,  false, b1q, (short)0, acc1, false, false);
    acc1 = __builtin_amdgcn_wmma_f32_16x16x32_f16(false, a2, false, b1i, (short)0, acc1, false, false);
    acc2 = __builtin_amdgcn_wmma_f32_16x16x32_f16(false, a,  false, b2q, (short)0, acc2, false, false);
    acc2 = __builtin_amdgcn_wmma_f32_16x16x32_f16(false, a2, false, b2i, (short)0, acc2, false, false);
  }
  __syncthreads();  // f-tile dead; reuse LDS for S-tile

  // C/D layout: lane (m/8)*16 + n, vgpr r = m%8  ->  m = (lane>>4)*8 + r, n = lane&15
  {
    const int n = lane & 15;
    const int mbase = m0t * 16 + (lane >> 4) * 8;
    for (int r = 0; r < 8; ++r) {
      Stile[(mbase + r) * SPAD + (ng * 16 + n)]       = acc0[r];
      Stile[(mbase + r) * SPAD + ((ng + 1) * 16 + n)] = acc1[r];
      Stile[(mbase + r) * SPAD + ((ng + 2) * 16 + n)] = acc2[r];
    }
  }
  __syncthreads();

  // -- epilogue: log_prob = -0.5*S + bias ; max over M ; LayerNorm over K --
  if (tid < 32) {
    float mk[KCLS];
    for (int k = 0; k < KCLS; ++k) {
      float best = -3.0e38f;
      for (int m = 0; m < MCMP; ++m) {
        int p = k * MCMP + m;
        float lp = -0.5f * Stile[tid * SPAD + p] + bias[p];
        best = fmaxf(best, lp);
      }
      mk[k] = best;
    }
    float su = 0.0f;
    for (int k = 0; k < KCLS; ++k) su += mk[k];
    float mean = su * (1.0f / (float)KCLS);
    float vq = 0.0f;
    for (int k = 0; k < KCLS; ++k) { float d = mk[k] - mean; vq += d * d; }
    float is = rsqrtf(vq * (1.0f / (float)KCLS) + 1e-5f);
    float* op = out + (size_t)b * KCLS * HWn + hw + tid;  // lane == px -> coalesced per k
    for (int k = 0; k < KCLS; ++k) {
      op[(size_t)k * HWn] = (mk[k] - mean) * is * mgamma[k] + mbeta[k];
    }
  }
}

// ---------------------------------------------------------------------------
extern "C" void kernel_launch(void* const* d_in, const int* in_sizes, int n_in,
                              void* d_out, int out_size, void* d_ws, size_t ws_size,
                              hipStream_t stream) {
  (void)in_sizes; (void)n_in; (void)out_size; (void)ws_size;
  const float* x      = (const float*)d_in[0];
  const float* means  = (const float*)d_in[1];
  const float* diag   = (const float*)d_in[2];
  const float* fgamma = (const float*)d_in[3];
  const float* fbeta  = (const float*)d_in[4];
  const float* mgamma = (const float*)d_in[5];
  const float* mbeta  = (const float*)d_in[6];
  float* out = (float*)d_out;

  char* ws = (char*)d_ws;                          // needs ~197 KB
  _Float16* Wq2 = (_Float16*)ws;                                    // 96*512 f16
  _Float16* Wiv = (_Float16*)(ws + (size_t)PPAD * C_DIM * 2);       // 96*512 f16
  float*    bias = (float*)(ws + (size_t)2 * PPAD * C_DIM * 2);     // 96 f32

  gmm_prep<<<PPAD, 128, 0, stream>>>(means, diag, Wq2, Wiv, bias);
  const int nblocks = (4 * HWn) / TPX;  // 65536 pixels / 32 = 2048
  gmm_main<<<nblocks, 128, 0, stream>>>(x, fgamma, fbeta, mgamma, mbeta,
                                        Wq2, Wiv, bias, out);
}
